// MemNet_17961553232139
// MI455X (gfx1250) — matrix operand order repaired and checked by reading, not compiled, not run on hardware
//
#include <hip/hip_runtime.h>
#include <hip/hip_bf16.h>
#include <math.h>
#include <stdint.h>

// ---------------- model dims ----------------
#define B_     4
#define T_     128
#define D_     256
#define FF_    1024
#define L_     2
#define NH_    8
#define HDA_   32
#define VOC_   32000
#define MD_    128
#define MH_    4
#define HD_    32
#define SLOTS_ 512
#define ROWS_  (B_*T_)            // 512
#define IFW_   (MH_*(4*HD_+1))    // 516

// ---------------- vector types ----------------
typedef __attribute__((ext_vector_type(8)))  float          v8f;
typedef __attribute__((ext_vector_type(16))) __bf16         v16bf;
typedef __attribute__((ext_vector_type(8)))  unsigned short usv8;
typedef __attribute__((ext_vector_type(16))) unsigned short usv16;

#if defined(__has_builtin)
#if __has_builtin(__builtin_amdgcn_tensor_load_to_lds)
#define HAVE_TDM 1
#endif
#endif
#ifndef HAVE_TDM
#define HAVE_TDM 0
#endif

// ---------------- helpers ----------------
__device__ inline unsigned short f2bf(float f) {
  unsigned int u = __float_as_uint(f);
  u += 0x7fffu + ((u >> 16) & 1u);          // round-to-nearest-even
  return (unsigned short)(u >> 16);
}

__device__ inline v8f wmma_bf16(v16bf a, v16bf b, v8f c) {
  // D = A(16x32 bf16) * B(32x16 bf16) + C(16x16 f32)
  return __builtin_amdgcn_wmma_f32_16x16x32_bf16(false, a, false, b, (short)0, c,
                                                 false, false);
}

// A-fragment: lane holds 16 consecutive halves at p[0..15]
__device__ inline v16bf ldfragA(const unsigned short* p) {
  union { usv16 v; usv8 h[2]; } t;
  t.h[0] = *(const usv8*)(p);
  t.h[1] = *(const usv8*)(p + 8);
  return __builtin_bit_cast(v16bf, t.v);
}
// B-fragment: lane holds 16 consecutive halves
__device__ inline v16bf ldfragB(const unsigned short* p) {
  return __builtin_bit_cast(v16bf, *(const usv16*)p);
}

// CDNA5 async copy: 16 bytes global -> LDS, tracked by ASYNCcnt.
// saddr form: per-lane 32-bit byte offset + wave-uniform 64-bit base.
__device__ inline void async_cp16(uint32_t lds_addr, const void* base, uint32_t byte_off) {
  asm volatile("global_load_async_to_lds_b128 %0, %1, %2"
               :: "v"(lds_addr), "v"(byte_off), "s"(base) : "memory");
}

#if HAVE_TDM
typedef __attribute__((ext_vector_type(4))) unsigned int v4u_;
typedef __attribute__((ext_vector_type(8))) int          v8i_;
typedef __attribute__((ext_vector_type(4))) int          v4i_;

// CDNA5 Tensor Data Mover: strided 2D tile (rows x 32 bf16 elems, row stride Kel)
// from global into contiguous LDS.  D# per CDNA5 ISA.
// This toolchain exposes the 6-arg builtin form:
//   (uint32x4 g0, int32x8 g1, int32x4 g2, int32x4 g3, int32x8 g4, i32 cpol)
__device__ inline void tdm_load_tile_b(const void* gbase, size_t byte_off,
                                       uint32_t lds_addr, int Kel, int rows) {
  uint64_t ga = (uint64_t)(uintptr_t)gbase + (uint64_t)byte_off;
  v4u_ g0;
  g0[0] = 1u;                                                  // count=1 (valid D#)
  g0[1] = lds_addr;                                            // LDS dest byte addr
  g0[2] = (unsigned)ga;                                        // global_addr[31:0]
  g0[3] = (unsigned)((ga >> 32) & 0x1ffffffULL) | (2u << 30);  // addr[56:32] | type=2
  v8i_ g1;
  g1[0] = (int)(1u << 16);                     // wg_mask=0, data_size=1 (2 bytes)
  g1[1] = (int)(((unsigned)Kel & 0xffffu) << 16);       // tensor_dim0[15:0]
  g1[2] = (int)(((unsigned)Kel >> 16) & 0xffffu);       // tensor_dim0[31:16], dim1 lo16=0
  g1[3] = (int)(16u | (32u << 16));            // tensor_dim1=1<<20 (hi16), tile_dim0=32
  g1[4] = (int)(unsigned)rows;                 // tile_dim1=rows, tile_dim2=0
  g1[5] = (int)(unsigned)Kel;                  // tensor_dim0_stride[31:0] (elements)
  g1[6] = 0;                                   // stride[47:32]=0, dim1_stride=0
  g1[7] = 0;
  v4i_ gz4 = {0, 0, 0, 0};
  v8i_ gz8 = {0, 0, 0, 0, 0, 0, 0, 0};
  __builtin_amdgcn_tensor_load_to_lds(g0, g1, gz4, gz4, gz8, 0);
}
#endif

__device__ inline float gelu_t(float x) {   // tanh approximation (jax.nn.gelu default)
  float x3 = x * x * x;
  return 0.5f * x * (1.f + tanhf(0.7978845608028654f * (x + 0.044715f * x3)));
}
__device__ inline float sigm(float x) { return 1.f / (1.f + __expf(-x)); }

// ---------------- small kernels ----------------
__global__ __launch_bounds__(256) void cvt_bf16_k(const float* __restrict__ in,
                                                  unsigned short* __restrict__ out,
                                                  int n) {
  int i = blockIdx.x * 256 + threadIdx.x;
  if (i < n) out[i] = f2bf(in[i]);
}

// weight convert WITH transpose: in[K][N] f32 -> out[N][K] bf16
__global__ __launch_bounds__(256) void cvtT_bf16_k(const float* __restrict__ in,
                                                   unsigned short* __restrict__ out,
                                                   int K, int N) {
  int i = blockIdx.x * 256 + threadIdx.x;
  if (i >= K * N) return;
  int n = i / K, k = i - n * K;
  out[i] = f2bf(in[(size_t)k * N + n]);
}

__global__ __launch_bounds__(256) void embed_k(const int* __restrict__ ids,
                                               const float* __restrict__ tok,
                                               const float* __restrict__ pos,
                                               float* __restrict__ X) {
  int i = blockIdx.x * 256 + threadIdx.x;
  if (i >= ROWS_ * D_) return;
  int rt = i >> 8;          // row = b*T + t
  int d  = i & (D_ - 1);
  int t  = rt & (T_ - 1);
  X[i] = tok[(size_t)ids[rt] * D_ + d] + pos[(size_t)t * D_ + d];
}

__global__ __launch_bounds__(256) void concat_bf16_k(const float* __restrict__ H,
                                                     const float* __restrict__ RV,
                                                     unsigned short* __restrict__ out) {
  int i = blockIdx.x * 256 + threadIdx.x;
  if (i >= ROWS_ * (D_ + MD_)) return;
  int r = i / (D_ + MD_);
  int c = i - r * (D_ + MD_);
  float v = (c < D_) ? H[(size_t)r * D_ + c] : RV[(size_t)r * MD_ + (c - D_)];
  out[i] = f2bf(v);
}

__global__ __launch_bounds__(256) void layernorm_k(const float* __restrict__ X,
                                                   const float* __restrict__ g,
                                                   const float* __restrict__ bb,
                                                   float* __restrict__ O) {
  const int row = blockIdx.x, tid = threadIdx.x;   // 256 threads, D_=256
  float v = X[(size_t)row * D_ + tid];
  float s = v, q = v * v;
  for (int o = 16; o > 0; o >>= 1) {
    s += __shfl_xor(s, o, 32);
    q += __shfl_xor(q, o, 32);
  }
  __shared__ float rs[8], rq[8];
  const int w = tid >> 5, lane = tid & 31;
  if (lane == 0) { rs[w] = s; rq[w] = q; }
  __syncthreads();
  if (tid == 0) {
    float ts = 0.f, tq = 0.f;
    for (int i = 0; i < 8; ++i) { ts += rs[i]; tq += rq[i]; }
    rs[0] = ts; rq[0] = tq;
  }
  __syncthreads();
  float mean = rs[0] * (1.f / D_);
  float var  = rq[0] * (1.f / D_) - mean * mean;
  O[(size_t)row * D_ + tid] = (v - mean) * rsqrtf(var + 1e-5f) * g[tid] + bb[tid];
}

// ---------------- WMMA GEMM: TDM weights + async activations, double buffered ----
// C[M,N] = epi( A[M,K](bf16 row-major) @ Bt[N,K](bf16, pre-transposed weights)^T )
// EPI: bit0 = +bias[N], bit1 = gelu, bit2 = +R[M,N] residual, bit3 = NT store
// Block tile 64(M) x 128(N), K-step 32, 256 threads = 8 waves.
template <int EPI>
__global__ __launch_bounds__(256) void gemm_bf16_wmma(
    const unsigned short* __restrict__ A, const unsigned short* __restrict__ Bt,
    const float* __restrict__ bias, const float* __restrict__ R,
    float* __restrict__ C, int N, int K) {
  __shared__ __align__(16) unsigned short As[2][64 * 32];    // [m][k]  2x4KB
  __shared__ __align__(16) unsigned short Bs[2][128 * 32];   // [n][k]  2x8KB
  const int tid  = threadIdx.x;
  const int lane = tid & 31, w = tid >> 5;
  const int wy = w & 3, wx = w >> 2;
  const int r15 = lane & 15, hi = lane >> 4;
  const int bm = blockIdx.y * 64, bn = blockIdx.x * 128;

  const int ar = tid >> 2, ac = (tid & 3) * 8;   // A tile: 64x32 = 256 x 16B chunks

  auto issueA = [&](int k0, int buf) {
    async_cp16((uint32_t)(uintptr_t)&As[buf][ar * 32 + ac], A,
               (uint32_t)(((size_t)(bm + ar) * K + k0 + ac) * 2));
  };
  auto issueB = [&](int k0, int buf) {
#if HAVE_TDM
    if (w == 0)   // one TDM descriptor moves the whole 128x32 strided tile
      tdm_load_tile_b(Bt, ((size_t)bn * K + k0) * 2,
                      (uint32_t)(uintptr_t)&Bs[buf][0], K, 128);
#else
#pragma unroll
    for (int s = 0; s < 2; ++s) {
      int j = tid + s * 256;
      int br = j >> 2, bc = (j & 3) * 8;
      async_cp16((uint32_t)(uintptr_t)&Bs[buf][br * 32 + bc], Bt,
                 (uint32_t)(((size_t)(bn + br) * K + k0 + bc) * 2));
    }
#endif
  };

  v8f acc[4] = {{}, {}, {}, {}};
  issueA(0, 0); issueB(0, 0);
  const int nk = K >> 5;
  for (int it = 0; it < nk; ++it) {
    const int cur = it & 1;
    if (it + 1 < nk) {
      issueA((it + 1) << 5, cur ^ 1);
      issueB((it + 1) << 5, cur ^ 1);
#if HAVE_TDM
      asm volatile("s_wait_asynccnt 0x1" ::: "memory");
      if (w == 0) __builtin_amdgcn_s_wait_tensorcnt(1);
#else
      asm volatile("s_wait_asynccnt 0x3" ::: "memory");
#endif
    } else {
      asm volatile("s_wait_asynccnt 0x0" ::: "memory");
#if HAVE_TDM
      if (w == 0) __builtin_amdgcn_s_wait_tensorcnt(0);
#endif
    }
    __syncthreads();
    v16bf af = ldfragA(&As[cur][(wy * 16 + r15) * 32 + hi * 16]);
#pragma unroll
    for (int nb = 0; nb < 4; ++nb) {
      v16bf bf = ldfragB(&Bs[cur][(wx * 64 + nb * 16 + r15) * 32 + hi * 16]);
      acc[nb] = wmma_bf16(af, bf, acc[nb]);
    }
    __syncthreads();
  }

#pragma unroll
  for (int nb = 0; nb < 4; ++nb) {
    const int colBase = bn + wx * 64 + nb * 16 + r15;
#pragma unroll
    for (int i = 0; i < 8; ++i) {
      const int row = bm + wy * 16 + i + hi * 8;   // C layout: M = i + 8*(lane>=16)
      float v = acc[nb][i];
      if constexpr ((EPI & 1) != 0) v += bias[colBase];
      if constexpr ((EPI & 2) != 0) v = gelu_t(v);
      if constexpr ((EPI & 4) != 0) v += R[(size_t)row * N + colBase];
      if constexpr ((EPI & 8) != 0)
        __builtin_nontemporal_store(v, &C[(size_t)row * N + colBase]);
      else
        C[(size_t)row * N + colBase] = v;
    }
  }
}

// ---------------- WMMA attention: one workgroup per (b, head) ----------------
__global__ __launch_bounds__(256) void attention_k(const float* __restrict__ QKV,
                                                   float* __restrict__ ATT) {
  __shared__ __align__(16) unsigned short Qs[T_ * HDA_];  // [t][d] 8KB
  __shared__ __align__(16) unsigned short Ks[T_ * HDA_];  // [t][d] 8KB
  __shared__ __align__(16) unsigned short Vt[HDA_ * T_];  // [d][t] 8KB
  __shared__ __align__(16) unsigned short Ps[T_ * T_];    // [q][k] 32KB
  const int bh = blockIdx.x;
  const int b = bh >> 3, hh = bh & 7;
  const int tid = threadIdx.x;
  for (int i = tid; i < T_ * HDA_; i += 256) {
    int t = i >> 5, d = i & 31;
    const float* base = QKV + (size_t)(b * T_ + t) * (3 * D_) + hh * HDA_;
    Qs[t * HDA_ + d] = f2bf(base[d]);
    Ks[t * HDA_ + d] = f2bf(base[D_ + d]);
    Vt[d * T_ + t]   = f2bf(base[2 * D_ + d]);
  }
  __syncthreads();
  const int w = tid >> 5, lane = tid & 31;
  const int r15 = lane & 15, hi = lane >> 4;

  v16bf qf = ldfragA(&Qs[(w * 16 + r15) * HDA_ + hi * 16]);
  float scv[8][8];
  const float scale = 0.1767766952966369f;  // 1/sqrt(32)
#pragma unroll
  for (int nb = 0; nb < 8; ++nb) {
    v16bf kf = ldfragB(&Ks[(nb * 16 + r15) * HDA_ + hi * 16]);
    v8f az = {};
    az = wmma_bf16(qf, kf, az);
#pragma unroll
    for (int i = 0; i < 8; ++i) {
      int qi = w * 16 + i + hi * 8;
      int ki = nb * 16 + r15;
      scv[nb][i] = (ki <= qi) ? az[i] * scale : -3.0e38f;   // causal mask
    }
  }
#pragma unroll
  for (int i = 0; i < 8; ++i) {
    float mx = -3.0e38f;
#pragma unroll
    for (int nb = 0; nb < 8; ++nb) mx = fmaxf(mx, scv[nb][i]);
    for (int o = 8; o > 0; o >>= 1) mx = fmaxf(mx, __shfl_xor(mx, o, 32));
    float sum = 0.f;
#pragma unroll
    for (int nb = 0; nb < 8; ++nb) {
      float p = __expf(scv[nb][i] - mx);
      scv[nb][i] = p; sum += p;
    }
    for (int o = 8; o > 0; o >>= 1) sum += __shfl_xor(sum, o, 32);
    float inv = 1.f / sum;
    int q = w * 16 + i + hi * 8;
#pragma unroll
    for (int nb = 0; nb < 8; ++nb) Ps[q * T_ + nb * 16 + r15] = f2bf(scv[nb][i] * inv);
  }
  __syncthreads();
  v8f o0 = {}; v8f o1 = {};
#pragma unroll
  for (int ks = 0; ks < 4; ++ks) {
    v16bf pf = ldfragA(&Ps[(w * 16 + r15) * T_ + ks * 32 + hi * 16]);
    v16bf v0 = ldfragB(&Vt[r15 * T_ + ks * 32 + hi * 16]);
    v16bf v1 = ldfragB(&Vt[(16 + r15) * T_ + ks * 32 + hi * 16]);
    o0 = wmma_bf16(pf, v0, o0);
    o1 = wmma_bf16(pf, v1, o1);
  }
#pragma unroll
  for (int i = 0; i < 8; ++i) {
    int q = w * 16 + i + hi * 8;
    ATT[(size_t)(b * T_ + q) * D_ + hh * HDA_ + r15]      = o0[i];
    ATT[(size_t)(b * T_ + q) * D_ + hh * HDA_ + 16 + r15] = o1[i];
  }
}

// ---------------- sequential memory scan (M kept in 320KB WGP LDS) ----------------
__device__ inline void topk8_softmax_512(float* sc, int lane) {
  float prev = 3.0e38f, m0 = 0.f;
  for (int it = 0; it < 8; ++it) {        // 8th-largest via repeated strict-max
    float lm = -3.0e38f;
    for (int j = 0; j < 16; ++j) {
      float v = sc[lane * 16 + j];
      if (v < prev && v > lm) lm = v;
    }
    for (int o = 16; o > 0; o >>= 1) lm = fmaxf(lm, __shfl_xor(lm, o, 32));
    if (it == 0) m0 = lm;
    prev = lm;
  }
  float thr = prev, s = 0.f;
  for (int j = 0; j < 16; ++j) {
    int i = lane * 16 + j;
    float v = sc[i];
    float p = (v >= thr) ? __expf(v - m0) : 0.f;
    sc[i] = p; s += p;
  }
  for (int o = 16; o > 0; o >>= 1) s += __shfl_xor(s, o, 32);
  float inv = 1.f / s;
  for (int j = 0; j < 16; ++j) sc[lane * 16 + j] *= inv;
}

__global__ __launch_bounds__(128) void memscan_k(
    const float* __restrict__ Xf, const float* __restrict__ Wif,
    const float* __restrict__ bif, const float* __restrict__ betaR,
    const float* __restrict__ betaW, float* __restrict__ RVout) {
  extern __shared__ float sm[];
  float* M   = sm;                        // MH*SLOTS*HD = 65536 f32 (256KB)
  float* rv  = M + MH_ * SLOTS_ * HD_;    // 128
  float* cb  = rv + MD_;                  // 384
  float* ifc = cb + (D_ + MD_);           // 520
  float* sc  = ifc + 520;                 // MH*SLOTS
  int*  actI = (int*)(sc + MH_ * SLOTS_); // MH*SLOTS
  int*  actC = actI + MH_ * SLOTS_;       // MH

  const int b = blockIdx.x;
  const int tid = threadIdx.x;            // 128 threads = 4 waves (one per head)
  const int w = tid >> 5, lane = tid & 31;

  for (int i = tid; i < MH_ * SLOTS_ * HD_; i += 128) M[i] = 0.f;
  for (int i = tid; i < MD_; i += 128) rv[i] = 0.f;
  float br = betaR[0]; br = fminf(fmaxf(log1pf(__expf(br)), 1.f), 20.f);
  float bw = betaW[0]; bw = fminf(fmaxf(log1pf(__expf(bw)), 1.f), 20.f);
  __syncthreads();

  for (int t = 0; t < T_; ++t) {
    const float* h = Xf + (size_t)(b * T_ + t) * D_;
    for (int i = tid; i < D_; i += 128) cb[i] = h[i];
    for (int i = tid; i < MD_; i += 128) {
      cb[D_ + i] = rv[i];
      RVout[(size_t)(b * T_ + t) * MD_ + i] = rv[i];   // rv entering step t
    }
    __syncthreads();
    for (int j = tid; j < IFW_; j += 128) {
      float a = bif[j];
      for (int k = 0; k < D_ + MD_; ++k) a += cb[k] * Wif[(size_t)k * IFW_ + j];
      ifc[j] = a;
    }
    __syncthreads();

    const float* rk  = ifc + w * (4 * HD_ + 1);
    const float* wk  = rk + HD_;
    const float* wvv = rk + 2 * HD_;
    const float* ers = rk + 3 * HD_;
    const float addg = sigm(rk[4 * HD_]);
    float* scH = sc + (size_t)w * SLOTS_;

    // ---- read: w_r = topk_softmax(beta_r * rk . M)
    for (int j = 0; j < 16; ++j) {
      int s = lane * 16 + j;
      const float* Mr = M + ((size_t)w * SLOTS_ + s) * HD_;
      float a = 0.f;
      for (int d = 0; d < HD_; ++d) a += rk[d] * Mr[d];
      scH[s] = a * br;
    }
    topk8_softmax_512(scH, lane);
    if (lane == 0) {
      int c = 0;
      for (int s = 0; s < SLOTS_; ++s)
        if (scH[s] > 0.f) actI[w * SLOTS_ + (c++)] = s;
      actC[w] = c;
    }
    {
      int cnt = actC[w];
      float acc = 0.f;
      for (int j = 0; j < cnt; ++j) {
        int s = actI[w * SLOTS_ + j];
        acc += scH[s] * M[((size_t)w * SLOTS_ + s) * HD_ + lane];
      }
      rv[w * HD_ + lane] += acc;
    }
    // ---- write: w_w = topk_softmax(beta_w * wk . M), sparse M update
    for (int j = 0; j < 16; ++j) {
      int s = lane * 16 + j;
      const float* Mr = M + ((size_t)w * SLOTS_ + s) * HD_;
      float a = 0.f;
      for (int d = 0; d < HD_; ++d) a += wk[d] * Mr[d];
      scH[s] = a * bw;
    }
    topk8_softmax_512(scH, lane);
    if (lane == 0) {
      int c = 0;
      for (int s = 0; s < SLOTS_; ++s)
        if (scH[s] > 0.f) actI[w * SLOTS_ + (c++)] = s;
      actC[w] = c;
    }
    {
      int cnt = actC[w];
      float er = sigm(ers[lane]);
      float wv = wvv[lane];
      for (int j = 0; j < cnt; ++j) {
        int s = actI[w * SLOTS_ + j];
        float* Mr = M + ((size_t)w * SLOTS_ + s) * HD_;
        float ww = scH[s];
        Mr[lane] = Mr[lane] * (1.f - ww * er) + addg * ww * wv;
      }
    }
    __syncthreads();
  }
}

// ---------------- workspace layout (bytes) ----------------
static constexpr size_t OFF_X     = 0;
static constexpr size_t OFF_H     = OFF_X     + (size_t)ROWS_ * D_ * 4;
static constexpr size_t OFF_QKV   = OFF_H     + (size_t)ROWS_ * D_ * 4;
static constexpr size_t OFF_ATT   = OFF_QKV   + (size_t)ROWS_ * 3 * D_ * 4;
static constexpr size_t OFF_FF1   = OFF_ATT   + (size_t)ROWS_ * D_ * 4;
static constexpr size_t OFF_RV    = OFF_FF1   + (size_t)ROWS_ * FF_ * 4;
static constexpr size_t OFF_ABF   = OFF_RV    + (size_t)ROWS_ * MD_ * 4;
static constexpr size_t OFF_WQKVB = OFF_ABF   + (size_t)ROWS_ * FF_ * 2;
static constexpr size_t OFF_WOB   = OFF_WQKVB + (size_t)L_ * D_ * 3 * D_ * 2;
static constexpr size_t OFF_W1B   = OFF_WOB   + (size_t)L_ * D_ * D_ * 2;
static constexpr size_t OFF_W2B   = OFF_W1B   + (size_t)L_ * D_ * FF_ * 2;
static constexpr size_t OFF_WLOGB = OFF_W2B   + (size_t)L_ * FF_ * D_ * 2;
// total ws need = OFF_WLOGB + (D_+MD_)*VOC_*2  ~= 34.3 MB

extern "C" void kernel_launch(void* const* d_in, const int* in_sizes, int n_in,
                              void* d_out, int out_size, void* d_ws, size_t ws_size,
                              hipStream_t stream) {
  (void)in_sizes; (void)n_in; (void)out_size; (void)ws_size;
  const int*   ids   = (const int*)  d_in[0];
  const float* tok   = (const float*)d_in[1];
  const float* pos   = (const float*)d_in[2];
  const float* Wqkv  = (const float*)d_in[3];
  const float* Wo    = (const float*)d_in[4];
  const float* ln1g  = (const float*)d_in[5];
  const float* ln1b  = (const float*)d_in[6];
  const float* ln2g  = (const float*)d_in[7];
  const float* ln2b  = (const float*)d_in[8];
  const float* W1    = (const float*)d_in[9];
  const float* b1    = (const float*)d_in[10];
  const float* W2    = (const float*)d_in[11];
  const float* b2    = (const float*)d_in[12];
  const float* lnfg  = (const float*)d_in[13];
  const float* lnfb  = (const float*)d_in[14];
  const float* Wlog  = (const float*)d_in[15];
  const float* blog  = (const float*)d_in[16];
  const float* Wif   = (const float*)d_in[17];
  const float* bif   = (const float*)d_in[18];
  const float* betaR = (const float*)d_in[19];
  const float* betaW = (const float*)d_in[20];
  float* out = (float*)d_out;

  char* ws = (char*)d_ws;
  float* X    = (float*)(ws + OFF_X);
  float* H    = (float*)(ws + OFF_H);
  float* QKV  = (float*)(ws + OFF_QKV);
  float* ATT  = (float*)(ws + OFF_ATT);
  float* FF1  = (float*)(ws + OFF_FF1);
  float* RV   = (float*)(ws + OFF_RV);
  unsigned short* Abf   = (unsigned short*)(ws + OFF_ABF);
  unsigned short* WqkvB = (unsigned short*)(ws + OFF_WQKVB);
  unsigned short* WoB   = (unsigned short*)(ws + OFF_WOB);
  unsigned short* W1B   = (unsigned short*)(ws + OFF_W1B);
  unsigned short* W2B   = (unsigned short*)(ws + OFF_W2B);
  unsigned short* WlogB = (unsigned short*)(ws + OFF_WLOGB);

  auto cvt = [&](const float* src, unsigned short* dst, int n) {
    cvt_bf16_k<<<(n + 255) / 256, 256, 0, stream>>>(src, dst, n);
  };
  auto cvtT = [&](const float* src, unsigned short* dst, int K, int N) {
    cvtT_bf16_k<<<(K * N + 255) / 256, 256, 0, stream>>>(src, dst, K, N);
  };

  // weight conversion to bf16 with transpose ([K][N] -> [N][K]) each call
  for (int l = 0; l < L_; ++l) {
    cvtT(Wqkv + (size_t)l * D_ * 3 * D_, WqkvB + (size_t)l * D_ * 3 * D_, D_, 3 * D_);
    cvtT(Wo   + (size_t)l * D_ * D_,     WoB   + (size_t)l * D_ * D_,     D_, D_);
    cvtT(W1   + (size_t)l * D_ * FF_,    W1B   + (size_t)l * D_ * FF_,    D_, FF_);
    cvtT(W2   + (size_t)l * FF_ * D_,    W2B   + (size_t)l * FF_ * D_,    FF_, D_);
  }
  cvtT(Wlog, WlogB, D_ + MD_, VOC_);

  embed_k<<<(ROWS_ * D_) / 256, 256, 0, stream>>>(ids, tok, pos, X);

  for (int l = 0; l < L_; ++l) {
    layernorm_k<<<ROWS_, 256, 0, stream>>>(X, ln1g + l * D_, ln1b + l * D_, H);
    cvt(H, Abf, ROWS_ * D_);
    gemm_bf16_wmma<0><<<dim3(3 * D_ / 128, ROWS_ / 64), 256, 0, stream>>>(
        Abf, WqkvB + (size_t)l * D_ * 3 * D_, nullptr, nullptr, QKV, 3 * D_, D_);
    attention_k<<<B_ * NH_, 256, 0, stream>>>(QKV, ATT);
    cvt(ATT, Abf, ROWS_ * D_);
    gemm_bf16_wmma<4><<<dim3(D_ / 128, ROWS_ / 64), 256, 0, stream>>>(
        Abf, WoB + (size_t)l * D_ * D_, nullptr, X, X, D_, D_);   // x += o @ Wo
    layernorm_k<<<ROWS_, 256, 0, stream>>>(X, ln2g + l * D_, ln2b + l * D_, H);
    cvt(H, Abf, ROWS_ * D_);
    gemm_bf16_wmma<3><<<dim3(FF_ / 128, ROWS_ / 64), 256, 0, stream>>>(
        Abf, W1B + (size_t)l * D_ * FF_, b1 + l * FF_, nullptr, FF1, FF_, D_);
    cvt(FF1, Abf, ROWS_ * FF_);
    gemm_bf16_wmma<5><<<dim3(D_ / 128, ROWS_ / 64), 256, 0, stream>>>(
        Abf, W2B + (size_t)l * FF_ * D_, b2 + l * D_, X, X, D_, FF_);
  }

  layernorm_k<<<ROWS_, 256, 0, stream>>>(X, lnfg, lnfb, H);

  // sequential memory scan: M (256KB) + state in dynamic LDS, one WG per batch
  size_t shmem = (size_t)(MH_ * SLOTS_ * HD_ + MD_ + (D_ + MD_) + 520 + MH_ * SLOTS_) * 4
               + (size_t)(MH_ * SLOTS_ + MH_) * 4;
  memscan_k<<<B_, 128, shmem, stream>>>(H, Wif, bif, betaR, betaW, RV);

  // logits = [H | RV] @ W_logits + b_logits  (one streaming pass; NT stores)
  concat_bf16_k<<<(ROWS_ * (D_ + MD_) + 255) / 256, 256, 0, stream>>>(H, RV, Abf);
  gemm_bf16_wmma<9><<<dim3(VOC_ / 128, ROWS_ / 64), 256, 0, stream>>>(
      Abf, WlogB, blog, nullptr, out, VOC_, D_ + MD_);
}
